// NetGCN_20469814132905
// MI455X (gfx1250) — matrix operand order repaired and checked by reading, not compiled, not run on hardware
//
#include <hip/hip_runtime.h>
#include <hip/hip_bf16.h>

typedef __bf16 bf16_t;
typedef __attribute__((ext_vector_type(16))) __bf16 v16bf;
typedef __attribute__((ext_vector_type(8)))  float  v8f;

#define N_NODES  50000
#define N_EDGES  800000
#define N_FEAT   128
#define DIM      64
#define N_GRAPHS 512

// out[M x DIM] = act(in[M x K]) @ W[K x DIM]; K in {64,128}; optional relu on input.
// One wave per 16-row tile; DIM=64 -> 4 accumulator tiles of 16x16 per wave.
// W is async-copied to LDS (fp32), then swizzled to bf16 so each lane's B
// fragment is one contiguous 32B LDS vector load.
template <int K, bool RELU_IN>
__global__ void gemm_wmma_bf16(const float* __restrict__ in,
                               const float* __restrict__ W,
                               float* __restrict__ out, int ntiles) {
  constexpr int NFRAG = (K / 32) * 8;                 // ksteps * 4 ntile * 2 half
  __shared__ __align__(16) float  stageW[K * DIM];    // raw fp32 W
  __shared__ __align__(32) bf16_t swzW[NFRAG * 256];  // fragment-major bf16

  // ---- async copy W (K*DIM fp32) into LDS via CDNA5 async-to-LDS path ----
  {
    const unsigned ldsBase = (unsigned)(size_t)(void*)stageW;
    const int bytesTotal = K * DIM * 4;
    const int stride = blockDim.x * 16;
    for (int off = threadIdx.x * 16; off < bytesTotal; off += stride) {
      unsigned ldsAddr = ldsBase + (unsigned)off;
      unsigned long long gaddr =
          (unsigned long long)(size_t)((const char*)W + off);
      asm volatile("global_load_async_to_lds_b128 %0, %1, off"
                   :: "v"(ldsAddr), "v"(gaddr) : "memory");
    }
    asm volatile("s_wait_asynccnt 0" ::: "memory");
  }
  __syncthreads();

  // ---- swizzle: frag fi=(kstep*8 + nt*2 + half); lane n holds elems e=0..15
  //      contiguous, where B elem e = W[kstep*32 + half*16 + e][nt*16 + n] ----
  for (int i = threadIdx.x; i < NFRAG * 256; i += blockDim.x) {
    const int fi = i >> 8;
    const int n  = (i >> 4) & 15;
    const int e  = i & 15;
    const int kstep = fi >> 3;
    const int nt    = (fi >> 1) & 3;
    const int half  = fi & 1;
    const int k   = kstep * 32 + half * 16 + e;
    const int col = nt * 16 + n;
    swzW[i] = (bf16_t)stageW[k * DIM + col];
  }
  __syncthreads();

  const int lane = threadIdx.x & 31;
  const int wave = threadIdx.x >> 5;
  const int tile = blockIdx.x * (blockDim.x >> 5) + wave;
  if (tile >= ntiles) return;

  const int row0 = tile << 4;
  const int m    = lane & 15;
  const int half = lane >> 4;

  v8f acc[4] = {};
  const float4* arow4 = (const float4*)(in + (size_t)(row0 + m) * K);

  for (int kstep = 0; kstep < K / 32; ++kstep) {
    // ---- A fragment (16x32 bf16, contiguous k = 16*half + e order, same as B), 4x b128 loads ----
    // half==0: elems 0..15 = K k0+0..15
    // half==1: elems 0..15 = K k0+16..31
    const int f0 = kstep * 8 + half * 4;
    float4 a0 = arow4[f0];
    float4 a1 = arow4[f0 + 1];
    float4 a2 = arow4[f0 + 2];
    float4 a3 = arow4[f0 + 3];
    if (RELU_IN) {
      a0.x = fmaxf(a0.x, 0.f); a0.y = fmaxf(a0.y, 0.f);
      a0.z = fmaxf(a0.z, 0.f); a0.w = fmaxf(a0.w, 0.f);
      a1.x = fmaxf(a1.x, 0.f); a1.y = fmaxf(a1.y, 0.f);
      a1.z = fmaxf(a1.z, 0.f); a1.w = fmaxf(a1.w, 0.f);
      a2.x = fmaxf(a2.x, 0.f); a2.y = fmaxf(a2.y, 0.f);
      a2.z = fmaxf(a2.z, 0.f); a2.w = fmaxf(a2.w, 0.f);
      a3.x = fmaxf(a3.x, 0.f); a3.y = fmaxf(a3.y, 0.f);
      a3.z = fmaxf(a3.z, 0.f); a3.w = fmaxf(a3.w, 0.f);
    }
    v16bf a;
    a[0]  = (bf16_t)a0.x; a[1]  = (bf16_t)a0.y;
    a[2]  = (bf16_t)a0.z; a[3]  = (bf16_t)a0.w;
    a[4]  = (bf16_t)a1.x; a[5]  = (bf16_t)a1.y;
    a[6]  = (bf16_t)a1.z; a[7]  = (bf16_t)a1.w;
    a[8]  = (bf16_t)a2.x; a[9]  = (bf16_t)a2.y;
    a[10] = (bf16_t)a2.z; a[11] = (bf16_t)a2.w;
    a[12] = (bf16_t)a3.x; a[13] = (bf16_t)a3.y;
    a[14] = (bf16_t)a3.z; a[15] = (bf16_t)a3.w;

#pragma unroll
    for (int nt = 0; nt < 4; ++nt) {
      const int fi = kstep * 8 + nt * 2 + half;
      const v16bf b = *(const v16bf*)(swzW + (fi << 8) + (m << 4));
      acc[nt] = __builtin_amdgcn_wmma_f32_16x16x32_bf16(
          false, a, false, b, (short)0, acc[nt], false, false);
    }
  }

  // C/D layout: VGPR j -> row (half*8 + j), col = nt*16 + (lane&15)
#pragma unroll
  for (int nt = 0; nt < 4; ++nt) {
#pragma unroll
    for (int j = 0; j < 8; ++j) {
      const int r = row0 + half * 8 + j;
      out[(size_t)r * DIM + (nt << 4) + m] = acc[nt][j];
    }
  }
}

// out[dst[e]][d] += in[src[e]][d]; 64 threads (2 waves) per edge, contiguous 256B.
__global__ void scatter_add_edges(const float* __restrict__ in,
                                  const int* __restrict__ src,
                                  const int* __restrict__ dst,
                                  float* __restrict__ out) {
  const long long gid = (long long)blockIdx.x * blockDim.x + threadIdx.x;
  const int e = (int)(gid >> 6);
  const int d = (int)(gid & 63);
  if (e >= N_EDGES) return;
  const int s = src[e];
  const int t = dst[e];
  unsafeAtomicAdd(out + (size_t)t * DIM + d, in[(size_t)s * DIM + d]);
}

// pooled[batch[n]][d] += h[n][d]; counts[batch[n]] += 1 (once per node)
__global__ void pool_sum(const float* __restrict__ h,
                         const int* __restrict__ batch,
                         float* __restrict__ pooled,
                         float* __restrict__ counts) {
  const long long gid = (long long)blockIdx.x * blockDim.x + threadIdx.x;
  const int n = (int)(gid >> 6);
  const int d = (int)(gid & 63);
  if (n >= N_NODES) return;
  const int g = batch[n];
  unsafeAtomicAdd(pooled + (size_t)g * DIM + d, h[(size_t)n * DIM + d]);
  if (d == 0) unsafeAtomicAdd(counts + g, 1.0f);
}

// out[g] = sigmoid( (pooled[g]/max(count,1)) . Wfc )
__global__ void head_fc_sigmoid(const float* __restrict__ pooled,
                                const float* __restrict__ counts,
                                const float* __restrict__ Wfc,
                                float* __restrict__ out) {
  const int g = blockIdx.x * blockDim.x + threadIdx.x;
  if (g >= N_GRAPHS) return;
  const float c = fmaxf(counts[g], 1.0f);
  float acc = 0.f;
#pragma unroll
  for (int d = 0; d < DIM; ++d)
    acc += pooled[(size_t)g * DIM + d] * Wfc[d];
  acc /= c;
  out[g] = 1.0f / (1.0f + __expf(-acc));
}

extern "C" void kernel_launch(void* const* d_in, const int* in_sizes, int n_in,
                              void* d_out, int out_size, void* d_ws, size_t ws_size,
                              hipStream_t stream) {
  (void)in_sizes; (void)n_in; (void)out_size; (void)ws_size;

  const float* x     = (const float*)d_in[0];  // [N_NODES, N_FEAT]
  const int*   edge  = (const int*)  d_in[1];  // [2, N_EDGES]
  const int*   batch = (const int*)  d_in[2];  // [N_NODES]
  const float* W1    = (const float*)d_in[3];  // [N_FEAT, DIM]
  const float* W2    = (const float*)d_in[4];  // [DIM, DIM]
  const float* Wfc   = (const float*)d_in[5];  // [DIM, 1]
  float*       out   = (float*)d_out;          // [N_GRAPHS, 1]

  float* t1     = (float*)d_ws;                       // x@W1        [N_NODES, DIM]
  float* h1     = t1 + (size_t)N_NODES * DIM;         // scatter out [N_NODES, DIM]
  float* t2     = h1 + (size_t)N_NODES * DIM;         // relu(h1)@W2 [N_NODES, DIM]
  float* h2     = t2 + (size_t)N_NODES * DIM;         // scatter out [N_NODES, DIM]
  float* pooled = h2 + (size_t)N_NODES * DIM;         // [N_GRAPHS, DIM]
  float* counts = pooled + (size_t)N_GRAPHS * DIM;    // [N_GRAPHS]

  const int* src = edge;
  const int* dst = edge + N_EDGES;

  hipMemsetAsync(h1, 0, sizeof(float) * (size_t)N_NODES * DIM, stream);
  hipMemsetAsync(h2, 0, sizeof(float) * (size_t)N_NODES * DIM, stream);
  hipMemsetAsync(pooled, 0, sizeof(float) * ((size_t)N_GRAPHS * DIM + N_GRAPHS), stream);

  const int ntiles = N_NODES / 16;            // 3125 (exact)
  const dim3 blk(256);                        // 8 waves/block
  const int gemm_blocks = (ntiles + 7) / 8;   // 391

  gemm_wmma_bf16<N_FEAT, false><<<gemm_blocks, blk, 0, stream>>>(x, W1, t1, ntiles);

  const long long ethreads = (long long)N_EDGES * 64;   // 51.2M
  scatter_add_edges<<<(int)(ethreads / 256), blk, 0, stream>>>(t1, src, dst, h1);

  gemm_wmma_bf16<DIM, true><<<gemm_blocks, blk, 0, stream>>>(h1, W2, t2, ntiles);

  scatter_add_edges<<<(int)(ethreads / 256), blk, 0, stream>>>(t2, src, dst, h2);

  const long long pthreads = (long long)N_NODES * 64;   // 3.2M
  pool_sum<<<(int)((pthreads + 255) / 256), blk, 0, stream>>>(h2, batch, pooled, counts);

  head_fc_sigmoid<<<(N_GRAPHS + 255) / 256, blk, 0, stream>>>(pooled, counts, Wfc, out);
}
